// NERRelationModel_54408645705883
// MI455X (gfx1250) — matrix operand order repaired and checked by reading, not compiled, not run on hardware
//
#include <hip/hip_runtime.h>

// ---------------- problem constants ----------------
#define Bc   48
#define Sc   512
#define Hc   768
#define Ec   128
#define Lc   16
#define Pc   256
#define NERc 11
#define FEATc 928   // 64 + 64 + 32 + 768

typedef __attribute__((ext_vector_type(16))) _Float16 v16h;
typedef __attribute__((ext_vector_type(8)))  _Float16 v8h;
typedef __attribute__((ext_vector_type(8)))  float    v8f;

// ---- fragment loaders (wave32 WMMA VGPR layouts, CDNA5 ISA) ----
__device__ __forceinline__ v16h load_a_frag(const _Float16* A, int ldA,
                                            int mbase, int kb, int lane) {
  int g = lane >> 4;
  int m = mbase + (lane & 15);
  const _Float16* p = A + (size_t)m * ldA + kb + 16 * g;
  v8h lo = *(const v8h*)p;          // K = kb+16g .. +7
  v8h hi = *(const v8h*)(p + 8);    // K = kb+16g+8 .. +7
  v16h a;
#pragma unroll
  for (int i = 0; i < 8; ++i) { a[i] = lo[i]; a[i + 8] = hi[i]; }
  return a;
}

__device__ __forceinline__ v16h load_b_frag(const _Float16* BT, int ldB,
                                            int nbase, int kb, int lane) {
  int g = lane >> 4;
  int n = nbase + (lane & 15);
  const _Float16* p = BT + (size_t)n * ldB + kb + 16 * g;
  v8h lo = *(const v8h*)p;
  v8h hi = *(const v8h*)(p + 8);
  v16h b;
#pragma unroll
  for (int i = 0; i < 8; ++i) { b[i] = lo[i]; b[i + 8] = hi[i]; }
  return b;
}

__device__ __forceinline__ v8f wmma_f16(v16h a, v16h b, v8f c) {
  return __builtin_amdgcn_wmma_f32_16x16x32_f16(false, a, false, b,
                                                (short)0, c, false, false);
}

// ====================================================================
// LDS-staged, async double-buffered WMMA GEMM.
//   C[m,n] = sum_k A[m,k] * BT[n,k]   (A: MxK f16, BT: NxK f16)
// Block = 128 threads (4 waves). Each wave owns one 16-row M tile;
// all waves share a 64-col x 32-k B slab staged into LDS with
// global_load_async_to_lds_b128 (ASYNCcnt), double buffered so slab
// k+1 streams in while WMMAs consume slab k. A frags are software-
// pipelined in registers. Last K-step is peeled so the steady-state
// loop is branch-free. Requires M%64==0, N%64==0, K%32==0.
// ====================================================================
__global__ void __launch_bounds__(128)
wmma_gemm_lds_kernel(const _Float16* __restrict__ A, const _Float16* __restrict__ BT,
                     const float* __restrict__ bias, float* __restrict__ C,
                     _Float16* __restrict__ C16,
                     int K, int ldA, int ldB, int ldC,
                     int Nout, int relu,
                     long long strideA, long long strideB, long long strideC) {
  __shared__ _Float16 bslab[2][64 * 32];   // 8 KB
  const int tid  = threadIdx.x;
  const int wave = tid >> 5;
  const int lane = tid & 31;
  const int mbase  = blockIdx.x * 64 + wave * 16;
  const int nchunk = blockIdx.y * 64;
  const int bat    = blockIdx.z;
  const _Float16* Ab = A  + (long long)bat * strideA;
  const _Float16* Bb = BT + (long long)bat * strideB;
  const unsigned sbase = (unsigned)(uintptr_t)(&bslab[0][0]);
  const int nk = K / 32;

  // Cooperatively DMA one 64n x 32k slab (4 KB) into LDS buffer `buf`:
  // 256 16-byte chunks, 2 per thread, chunk c -> row c>>2, sub c&3.
  auto issue_slab = [&](int buf, int kb) {
#pragma unroll
    for (int q = 0; q < 2; ++q) {
      int c = tid + q * 128;
      const _Float16* g = Bb + (size_t)(nchunk + (c >> 2)) * ldB + kb + (c & 3) * 8;
      unsigned l = sbase + (unsigned)buf * (64 * 32 * 2) + (unsigned)c * 16;
      asm volatile("global_load_async_to_lds_b128 %0, %1, off"
                   :: "v"(l), "v"(g) : "memory");
    }
  };
  auto lds_bfrag = [&](int buf, int t) -> v16h {
    int g = lane >> 4, nl = lane & 15;
    const _Float16* p = &bslab[buf][(t * 16 + nl) * 32 + 16 * g];
    v8h lo = *(const v8h*)p;
    v8h hi = *(const v8h*)(p + 8);
    v16h b;
#pragma unroll
    for (int i = 0; i < 8; ++i) { b[i] = lo[i]; b[i + 8] = hi[i]; }
    return b;
  };

  v8f acc[4];
#pragma unroll
  for (int t = 0; t < 4; ++t) acc[t] = (v8f)(0.f);

  issue_slab(0, 0);
  v16h a_cur = load_a_frag(Ab, ldA, mbase, 0, lane);

  // steady state: next slab streams in (ASYNCcnt) while WMMAs run
  for (int kbi = 0; kbi + 1 < nk; ++kbi) {
    const int kb  = kbi * 32;
    const int buf = kbi & 1;
    issue_slab(buf ^ 1, kb + 32);
    asm volatile("s_wait_asynccnt 0x2" ::: "memory");  // this wave's slab landed
    __syncthreads();                                    // slab visible to all waves
    __builtin_prefetch(Ab + (size_t)(mbase + (lane & 15)) * ldA + kb + 64, 0, 1);
    v16h a_next = load_a_frag(Ab, ldA, mbase, kb + 32, lane);
    // load all four B fragments first so dscnt waits can be graduated
    v16h b0 = lds_bfrag(buf, 0);
    v16h b1 = lds_bfrag(buf, 1);
    v16h b2 = lds_bfrag(buf, 2);
    v16h b3 = lds_bfrag(buf, 3);
    acc[0] = wmma_f16(a_cur, b0, acc[0]);
    acc[1] = wmma_f16(a_cur, b1, acc[1]);
    acc[2] = wmma_f16(a_cur, b2, acc[2]);
    acc[3] = wmma_f16(a_cur, b3, acc[3]);
    __syncthreads();                                    // reads of buf done before reuse
    a_cur = a_next;
  }
  // peeled tail
  {
    const int buf = (nk - 1) & 1;
    asm volatile("s_wait_asynccnt 0x0" ::: "memory");
    __syncthreads();
    v16h b0 = lds_bfrag(buf, 0);
    v16h b1 = lds_bfrag(buf, 1);
    v16h b2 = lds_bfrag(buf, 2);
    v16h b3 = lds_bfrag(buf, 3);
    acc[0] = wmma_f16(a_cur, b0, acc[0]);
    acc[1] = wmma_f16(a_cur, b1, acc[1]);
    acc[2] = wmma_f16(a_cur, b2, acc[2]);
    acc[3] = wmma_f16(a_cur, b3, acc[3]);
  }

  const int g  = lane >> 4;
  const int nl = lane & 15;
#pragma unroll
  for (int t = 0; t < 4; ++t) {
    int n = nchunk + 16 * t + nl;
    if (n >= Nout) continue;
    float bv = bias ? bias[n] : 0.f;
#pragma unroll
    for (int r = 0; r < 8; ++r) {
      int m = mbase + r + 8 * g;
      float v = acc[t][r] + bv;
      if (relu) v = v > 0.f ? v : 0.f;
      size_t idx = (size_t)m * ldC + n;
      if (C)   C[(long long)bat * strideC + idx] = v;
      if (C16) C16[idx] = (_Float16)v;
    }
  }
}

// ====================================================================
// Direct single-wave WMMA GEMM for skinny outputs (one 16-col tile).
// B fragments double-buffered in registers for graduated loadcnt waits.
// ====================================================================
__global__ void __launch_bounds__(32)
wmma_gemm_skinny_kernel(const _Float16* __restrict__ A, const _Float16* __restrict__ BT,
                        const float* __restrict__ bias, float* __restrict__ C,
                        int K, int ldA, int ldB, int ldC, int Nout) {
  const int lane  = threadIdx.x;
  const int mbase = blockIdx.x * 16;

  v8f acc = (v8f)(0.f);
  v16h a_cur = load_a_frag(A, ldA, mbase, 0, lane);
  v16h b_cur = load_b_frag(BT, ldB, 0, 0, lane);
  for (int kb = 0; kb < K; kb += 32) {
    v16h a_nxt = a_cur, b_nxt = b_cur;
    if (kb + 32 < K) {
      a_nxt = load_a_frag(A, ldA, mbase, kb + 32, lane);
      b_nxt = load_b_frag(BT, ldB, 0, kb + 32, lane);
    }
    acc = wmma_f16(a_cur, b_cur, acc);
    a_cur = a_nxt; b_cur = b_nxt;
  }

  const int g  = lane >> 4;
  const int nl = lane & 15;
  if (nl < Nout) {
    float bv = bias ? bias[nl] : 0.f;
#pragma unroll
    for (int r = 0; r < 8; ++r) {
      int m = mbase + r + 8 * g;
      C[(size_t)m * ldC + nl] = acc[r] + bv;
    }
  }
}

// ---------------- f32 -> f16 elementwise ----------------
__global__ void __launch_bounds__(256)
convert_f16_kernel(const float* __restrict__ in, _Float16* __restrict__ out, int n) {
  int i = blockIdx.x * 256 + threadIdx.x;
  if (i < n) out[i] = (_Float16)in[i];
}

// ------------- transpose+convert: in (K x N) f32 -> out (Npad x K) f16 -------------
__global__ void __launch_bounds__(256)
transpose_f16_kernel(const float* __restrict__ in, _Float16* __restrict__ out,
                     int K, int N) {
  int n = blockIdx.y;
  int k = blockIdx.x * 256 + threadIdx.x;
  if (k < K)
    out[(size_t)n * K + k] = (n < N) ? (_Float16)in[(size_t)k * N + n] : (_Float16)0.f;
}

// ------------- span pooling + type embedding -> x (f32 + f16) -------------
__global__ void __launch_bounds__(256)
span_pool_kernel(const float* __restrict__ seq, const int* __restrict__ span_start,
                 const int* __restrict__ span_len, const int* __restrict__ entity_type,
                 const float* __restrict__ type_emb,
                 float* __restrict__ x, _Float16* __restrict__ x16) {
  int e = blockIdx.x, b = blockIdx.y, t = threadIdx.x;
  int ss = span_start[b * Ec + e];
  int sl = span_len[b * Ec + e];
  int ty = entity_type[b * Ec + e];
  const float* base = seq + ((size_t)b * Sc + ss) * Hc;
  __shared__ float sc[Lc];
  float tv[3][Lc], mh[3];
  float inv_cnt = 1.f / (float)(sl + 1);
#pragma unroll
  for (int j = 0; j < 3; ++j) {
    int h = t + 256 * j;
    float s = 0.f;
#pragma unroll
    for (int l = 0; l < Lc; ++l) {
      float v = base[(size_t)l * Hc + h];
      tv[j][l] = v;
      if (l <= sl) s += v;
    }
    mh[j] = s * inv_cnt;
  }
  if (t < Lc) sc[t] = 0.f;
  __syncthreads();
#pragma unroll
  for (int l = 0; l < Lc; ++l) {
    float p = tv[0][l] * mh[0] + tv[1][l] * mh[1] + tv[2][l] * mh[2];
    atomicAdd(&sc[l], p);
  }
  __syncthreads();
  float mx = -3.4e38f;
  for (int l = 0; l <= sl; ++l) mx = fmaxf(mx, sc[l]);
  float w[Lc], sum = 0.f;
#pragma unroll
  for (int l = 0; l < Lc; ++l) {
    float wv = (l <= sl) ? __expf(sc[l] - mx) : 0.f;
    w[l] = wv; sum += wv;
  }
  float inv = 1.f / sum;
#pragma unroll
  for (int j = 0; j < 3; ++j) {
    int h = t + 256 * j;
    float pooled = 0.f;
#pragma unroll
    for (int l = 0; l < Lc; ++l) pooled += w[l] * tv[j][l];
    float xv = pooled * inv + type_emb[(size_t)ty * Hc + h];
    size_t idx = ((size_t)b * Ec + e) * Hc + h;
    x[idx] = xv;
    x16[idx] = (_Float16)xv;
  }
}

// ------------- row inverse norms of x -------------
__global__ void __launch_bounds__(256)
rownorm_kernel(const float* __restrict__ x, float* __restrict__ invn) {
  int row = blockIdx.x;
  float s = 0.f;
  for (int h = threadIdx.x; h < Hc; h += 256) {
    float v = x[(size_t)row * Hc + h];
    s += v * v;
  }
  __shared__ float red[256];
  red[threadIdx.x] = s;
  __syncthreads();
  for (int off = 128; off > 0; off >>= 1) {
    if (threadIdx.x < off) red[threadIdx.x] += red[threadIdx.x + off];
    __syncthreads();
  }
  if (threadIdx.x == 0) invn[row] = 1.f / fmaxf(sqrtf(red[0]), 1e-12f);
}

// ------------- top-6 neighbors + adjacency -------------
__global__ void __launch_bounds__(128)
topk_adj_kernel(const float* __restrict__ sim, const float* __restrict__ invn,
                unsigned char* __restrict__ adj) {
  int i = blockIdx.x, b = blockIdx.y, t = threadIdx.x;
  __shared__ float row[Ec];
  float invi = invn[b * Ec + i];
  row[t] = sim[((size_t)b * Ec + i) * Ec + t] * invi * invn[b * Ec + t];
  unsigned char* arow = adj + ((size_t)b * Ec + i) * Ec;
  arow[t] = (t == i) ? 1 : 0;
  __syncthreads();
  if (t == 0) {
    unsigned chosen[4] = {0u, 0u, 0u, 0u};
    for (int k = 0; k < 6; ++k) {
      float best = -3.4e38f; int bi = 0;
      for (int j = 0; j < Ec; ++j) {
        if ((chosen[j >> 5] >> (j & 31)) & 1u) continue;
        if (row[j] > best) { best = row[j]; bi = j; }
      }
      chosen[bi >> 5] |= 1u << (bi & 31);
      if (best > 0.1f && bi != i) arow[bi] = 1;
    }
  }
}

// ------------- GAT attention src/dst coefficients -------------
__global__ void __launch_bounds__(256)
attn_srcdst_kernel(const float* __restrict__ h, const float* __restrict__ a_src,
                   const float* __restrict__ a_dst, float* __restrict__ src,
                   float* __restrict__ dst, int heads, int fdim) {
  int row = blockIdx.x;
  int dim = heads * fdim;
  __shared__ float rs[256], rd[256];
  for (int hd = 0; hd < heads; ++hd) {
    float s = 0.f, d = 0.f;
    for (int f = threadIdx.x; f < fdim; f += 256) {
      float v = h[(size_t)row * dim + hd * fdim + f];
      s += v * a_src[hd * fdim + f];
      d += v * a_dst[hd * fdim + f];
    }
    rs[threadIdx.x] = s; rd[threadIdx.x] = d;
    __syncthreads();
    for (int off = 128; off > 0; off >>= 1) {
      if (threadIdx.x < off) {
        rs[threadIdx.x] += rs[threadIdx.x + off];
        rd[threadIdx.x] += rd[threadIdx.x + off];
      }
      __syncthreads();
    }
    if (threadIdx.x == 0) {
      src[(size_t)row * heads + hd] = rs[0];
      dst[(size_t)row * heads + hd] = rd[0];
    }
    __syncthreads();
  }
}

// ------------- GAT attention softmax over sources -------------
__global__ void __launch_bounds__(128)
attn_alpha_kernel(const float* __restrict__ src, const float* __restrict__ dst,
                  const unsigned char* __restrict__ adj, float* __restrict__ alpha,
                  int heads) {
  int tn = blockIdx.x, b = blockIdx.y, s = threadIdx.x;
  __shared__ float red[Ec];
  int a = adj[((size_t)b * Ec + s) * Ec + tn];   // A[b, s, t]
  for (int hd = 0; hd < heads; ++hd) {
    float ev = src[((size_t)b * Ec + s) * heads + hd] +
               dst[((size_t)b * Ec + tn) * heads + hd];
    ev = (ev > 0.f) ? ev : 0.2f * ev;  // leaky_relu(0.2)
    float masked = a ? ev : -3.4e38f;
    red[s] = masked; __syncthreads();
    for (int off = 64; off > 0; off >>= 1) {
      if (s < off) red[s] = fmaxf(red[s], red[s + off]);
      __syncthreads();
    }
    float mx = red[0]; __syncthreads();
    float ex = a ? __expf(ev - mx) : 0.f;
    red[s] = ex; __syncthreads();
    for (int off = 64; off > 0; off >>= 1) {
      if (s < off) red[s] += red[s + off];
      __syncthreads();
    }
    float denom = red[0]; __syncthreads();
    alpha[(((size_t)b * Ec + s) * Ec + tn) * heads + hd] = ex / denom;
  }
}

// ------------- aggregation + bias + LayerNorm + ELU -------------
__global__ void __launch_bounds__(256)
agg_ln_elu_kernel(const float* __restrict__ alpha, const float* __restrict__ h,
                  const float* __restrict__ bias, const float* __restrict__ gamma,
                  const float* __restrict__ beta, float* __restrict__ out,
                  _Float16* __restrict__ out16, int heads, int fdim) {
  int tn = blockIdx.x, b = blockIdx.y;
  int dim = heads * fdim;
  __shared__ float vals[512];
  __shared__ float red[256];
  for (int o = threadIdx.x; o < dim; o += 256) {
    int hd = o / fdim;
    float s = 0.f;
    for (int sn = 0; sn < Ec; ++sn) {
      s += alpha[(((size_t)b * Ec + sn) * Ec + tn) * heads + hd] *
           h[((size_t)b * Ec + sn) * dim + o];
    }
    vals[o] = s + bias[o];
  }
  __syncthreads();
  float part = 0.f;
  for (int o = threadIdx.x; o < dim; o += 256) part += vals[o];
  red[threadIdx.x] = part; __syncthreads();
  for (int off = 128; off > 0; off >>= 1) {
    if (threadIdx.x < off) red[threadIdx.x] += red[threadIdx.x + off];
    __syncthreads();
  }
  float mean = red[0] / (float)dim; __syncthreads();
  float vpart = 0.f;
  for (int o = threadIdx.x; o < dim; o += 256) {
    float dlt = vals[o] - mean; vpart += dlt * dlt;
  }
  red[threadIdx.x] = vpart; __syncthreads();
  for (int off = 128; off > 0; off >>= 1) {
    if (threadIdx.x < off) red[threadIdx.x] += red[threadIdx.x + off];
    __syncthreads();
  }
  float inv = rsqrtf(red[0] / (float)dim + 1e-5f);
  for (int o = threadIdx.x; o < dim; o += 256) {
    float v = (vals[o] - mean) * inv * gamma[o] + beta[o];
    v = (v > 0.f) ? v : (__expf(v) - 1.f);   // elu
    size_t idx = ((size_t)b * Ec + tn) * dim + o;
    out[idx] = v;
    if (out16) out16[idx] = (_Float16)v;
  }
}

// ------------- relation feature concat (f16) -------------
__global__ void __launch_bounds__(256)
feat_kernel(const float* __restrict__ ent, const float* __restrict__ seq,
            const float* __restrict__ rel_emb, const int* __restrict__ pair_head,
            const int* __restrict__ pair_tail, const int* __restrict__ pair_rel,
            _Float16* __restrict__ feat) {
  int p = blockIdx.x, b = blockIdx.y;
  int hidx = pair_head[b * Pc + p];
  int tidx = pair_tail[b * Pc + p];
  int r = pair_rel[b * Pc + p];
  _Float16* out = feat + ((size_t)b * Pc + p) * FEATc;
  for (int i = threadIdx.x; i < FEATc; i += 256) {
    float v;
    if (i < 64)        v = ent[((size_t)b * Ec + hidx) * 64 + i];
    else if (i < 128)  v = ent[((size_t)b * Ec + tidx) * 64 + (i - 64)];
    else if (i < 160)  v = rel_emb[(size_t)r * 32 + (i - 128)];
    else               v = seq[(size_t)b * Sc * Hc + (i - 160)];  // CLS token
    out[i] = (_Float16)v;
  }
}

// ====================================================================
extern "C" void kernel_launch(void* const* d_in, const int* in_sizes, int n_in,
                              void* d_out, int out_size, void* d_ws, size_t ws_size,
                              hipStream_t stream) {
  (void)in_sizes; (void)n_in; (void)out_size; (void)ws_size;
  const float* seq       = (const float*)d_in[0];
  const int*  span_start = (const int*)d_in[1];
  const int*  span_len   = (const int*)d_in[2];
  const int*  ent_type   = (const int*)d_in[3];
  const int*  pair_head  = (const int*)d_in[4];
  const int*  pair_tail  = (const int*)d_in[5];
  const int*  pair_rel   = (const int*)d_in[6];
  const float* W_ner1    = (const float*)d_in[7];
  const float* b_ner1    = (const float*)d_in[8];
  const float* W_ner2    = (const float*)d_in[9];
  const float* b_ner2    = (const float*)d_in[10];
  const float* type_emb  = (const float*)d_in[11];
  const float* W_g1      = (const float*)d_in[12];
  const float* a_src1    = (const float*)d_in[13];
  const float* a_dst1    = (const float*)d_in[14];
  const float* b_g1      = (const float*)d_in[15];
  const float* g1_gamma  = (const float*)d_in[16];
  const float* g1_beta   = (const float*)d_in[17];
  const float* W_g2      = (const float*)d_in[18];
  const float* a_src2    = (const float*)d_in[19];
  const float* a_dst2    = (const float*)d_in[20];
  const float* b_g2      = (const float*)d_in[21];
  const float* g2_gamma  = (const float*)d_in[22];
  const float* g2_beta   = (const float*)d_in[23];
  const float* rel_emb   = (const float*)d_in[24];
  const float* W_r1      = (const float*)d_in[25];
  const float* b_r1      = (const float*)d_in[26];
  const float* W_r2      = (const float*)d_in[27];
  const float* b_r2      = (const float*)d_in[28];

  float* ner_out = (float*)d_out;                           // (48,512,11)
  float* rel_out = (float*)d_out + (size_t)Bc * Sc * NERc;  // (48,256)

  char* ws = (char*)d_ws;
  size_t off = 0;
  auto alloc = [&](size_t bytes) -> char* {
    char* p = ws + off;
    off += (bytes + 255) & ~(size_t)255;
    return p;
  };
  _Float16* seq16   = (_Float16*)alloc((size_t)Bc * Sc * Hc * 2);
  _Float16* W1T     = (_Float16*)alloc((size_t)256 * Hc * 2);
  _Float16* W2T     = (_Float16*)alloc((size_t)16 * 256 * 2);
  _Float16* Wg1T    = (_Float16*)alloc((size_t)512 * Hc * 2);
  _Float16* Wg2T    = (_Float16*)alloc((size_t)64 * 512 * 2);
  _Float16* Wr1T    = (_Float16*)alloc((size_t)256 * FEATc * 2);
  _Float16* Wr2T    = (_Float16*)alloc((size_t)16 * 256 * 2);
  _Float16* hid1_16 = (_Float16*)alloc((size_t)Bc * Sc * 256 * 2);
  float*    x       = (float*)alloc((size_t)Bc * Ec * Hc * 4);
  _Float16* x16     = (_Float16*)alloc((size_t)Bc * Ec * Hc * 2);
  float*    invn    = (float*)alloc((size_t)Bc * Ec * 4);
  float*    sim     = (float*)alloc((size_t)Bc * Ec * Ec * 4);
  unsigned char* adj = (unsigned char*)alloc((size_t)Bc * Ec * Ec);
  float*    h1      = (float*)alloc((size_t)Bc * Ec * 512 * 4);
  float*    src1    = (float*)alloc((size_t)Bc * Ec * 4 * 4);
  float*    dst1    = (float*)alloc((size_t)Bc * Ec * 4 * 4);
  float*    alpha1  = (float*)alloc((size_t)Bc * Ec * Ec * 4 * 4);
  float*    g1      = (float*)alloc((size_t)Bc * Ec * 512 * 4);
  _Float16* g1_16   = (_Float16*)alloc((size_t)Bc * Ec * 512 * 2);
  float*    h2      = (float*)alloc((size_t)Bc * Ec * 64 * 4);
  float*    src2    = (float*)alloc((size_t)Bc * Ec * 4);
  float*    dst2    = (float*)alloc((size_t)Bc * Ec * 4);
  float*    alpha2  = (float*)alloc((size_t)Bc * Ec * Ec * 4);
  float*    ent     = (float*)alloc((size_t)Bc * Ec * 64 * 4);
  _Float16* feat16  = (_Float16*)alloc((size_t)Bc * Pc * FEATc * 2);
  _Float16* hid2_16 = (_Float16*)alloc((size_t)Bc * Pc * 256 * 2);

  // ---- f16 copy of sequence_output ----
  {
    int n = Bc * Sc * Hc;
    convert_f16_kernel<<<dim3((n + 255) / 256), dim3(256), 0, stream>>>(seq, seq16, n);
  }
  // ---- weight transposes to (Npad x K) f16 ----
  transpose_f16_kernel<<<dim3(3, 256), 256, 0, stream>>>(W_ner1, W1T, Hc, 256);
  transpose_f16_kernel<<<dim3(1, 16), 256, 0, stream>>>(W_ner2, W2T, 256, NERc);
  transpose_f16_kernel<<<dim3(3, 512), 256, 0, stream>>>(W_g1, Wg1T, Hc, 512);
  transpose_f16_kernel<<<dim3(2, 64), 256, 0, stream>>>(W_g2, Wg2T, 512, 64);
  transpose_f16_kernel<<<dim3(4, 256), 256, 0, stream>>>(W_r1, Wr1T, FEATc, 256);
  transpose_f16_kernel<<<dim3(1, 16), 256, 0, stream>>>(W_r2, Wr2T, 256, 1);

  // ---- NER MLP: hid = relu(seq @ W1 + b1);  ner = hid @ W2 + b2 ----
  wmma_gemm_lds_kernel<<<dim3(Bc * Sc / 64, 256 / 64, 1), 128, 0, stream>>>(
      seq16, W1T, b_ner1, nullptr, hid1_16,
      Hc, Hc, Hc, 256, 256, 1, 0, 0, 0);
  wmma_gemm_skinny_kernel<<<dim3(Bc * Sc / 16), 32, 0, stream>>>(
      hid1_16, W2T, b_ner2, ner_out, 256, 256, 256, NERc, NERc);

  // ---- span pooling + type embedding ----
  span_pool_kernel<<<dim3(Ec, Bc), 256, 0, stream>>>(
      seq, span_start, span_len, ent_type, type_emb, x, x16);

  // ---- similarity graph ----
  rownorm_kernel<<<dim3(Bc * Ec), 256, 0, stream>>>(x, invn);
  wmma_gemm_lds_kernel<<<dim3(Ec / 64, Ec / 64, Bc), 128, 0, stream>>>(
      x16, x16, nullptr, sim, nullptr,
      Hc, Hc, Hc, Ec, Ec, 0,
      (long long)Ec * Hc, (long long)Ec * Hc, (long long)Ec * Ec);
  topk_adj_kernel<<<dim3(Ec, Bc), 128, 0, stream>>>(sim, invn, adj);

  // ---- GAT layer 1 (4 heads x 128, concat) ----
  wmma_gemm_lds_kernel<<<dim3(Bc * Ec / 64, 512 / 64, 1), 128, 0, stream>>>(
      x16, Wg1T, nullptr, h1, nullptr,
      Hc, Hc, Hc, 512, 512, 0, 0, 0, 0);
  attn_srcdst_kernel<<<dim3(Bc * Ec), 256, 0, stream>>>(h1, a_src1, a_dst1, src1, dst1, 4, 128);
  attn_alpha_kernel<<<dim3(Ec, Bc), 128, 0, stream>>>(src1, dst1, adj, alpha1, 4);
  agg_ln_elu_kernel<<<dim3(Ec, Bc), 256, 0, stream>>>(
      alpha1, h1, b_g1, g1_gamma, g1_beta, g1, g1_16, 4, 128);

  // ---- GAT layer 2 (1 head x 64) ----
  wmma_gemm_lds_kernel<<<dim3(Bc * Ec / 64, 64 / 64, 1), 128, 0, stream>>>(
      g1_16, Wg2T, nullptr, h2, nullptr,
      512, 512, 512, 64, 64, 0, 0, 0, 0);
  attn_srcdst_kernel<<<dim3(Bc * Ec), 256, 0, stream>>>(h2, a_src2, a_dst2, src2, dst2, 1, 64);
  attn_alpha_kernel<<<dim3(Ec, Bc), 128, 0, stream>>>(src2, dst2, adj, alpha2, 1);
  agg_ln_elu_kernel<<<dim3(Ec, Bc), 256, 0, stream>>>(
      alpha2, h2, b_g2, g2_gamma, g2_beta, ent, nullptr, 1, 64);

  // ---- relation head ----
  feat_kernel<<<dim3(Pc, Bc), 256, 0, stream>>>(
      ent, seq, rel_emb, pair_head, pair_tail, pair_rel, feat16);
  wmma_gemm_lds_kernel<<<dim3(Bc * Pc / 64, 256 / 64, 1), 128, 0, stream>>>(
      feat16, Wr1T, b_r1, nullptr, hid2_16,
      FEATc, FEATc, FEATc, 256, 256, 1, 0, 0, 0);
  wmma_gemm_skinny_kernel<<<dim3(Bc * Pc / 16), 32, 0, stream>>>(
      hid2_16, Wr2T, b_r2, rel_out, 256, 256, 256, 1, 1);
}
